// ELPH_modified_66357244723795
// MI455X (gfx1250) — compile-verified, matrix operand order checked by reading, not run on hardware
//
#include <hip/hip_runtime.h>
#include <stdint.h>

// ---------------- problem constants (match reference) ----------------
#define N_NODES 20000
#define N_EDGES 320000
#define NE_TOT  (N_EDGES + N_NODES)   // + self loops
#define CH      128                   // IN_CH == HID == OUT_CH
#define NPERM   128
#define HLL_M   256
#define LABEL_DIM 8
// ALPHA * m * m computed in double, used as f32 (matches jax weak-typing)
#define AMM_CONST ((float)((0.7213 / (1.0 + 1.079 / 256.0)) * 256.0 * 256.0))

typedef __attribute__((ext_vector_type(16))) _Float16 v16h;
typedef __attribute__((ext_vector_type(8)))  float    v8f;

// ---------------- helpers ----------------
__device__ __forceinline__ unsigned bmax4(unsigned a, unsigned b) {
    unsigned r = 0u;
#pragma unroll
    for (int j = 0; j < 4; j++) {
        unsigned x = (a >> (8 * j)) & 255u, y = (b >> (8 * j)) & 255u;
        unsigned m = x > y ? x : y;
        r |= m << (8 * j);
    }
    return r;
}

__device__ __forceinline__ float hll_count_f(float sumExp, int zeros) {
    float e_raw = AMM_CONST / sumExp;
    float v = (float)zeros;
    float lc = 256.0f * logf(256.0f / fmaxf(v, 1.0f));   // linear counting
    return (zeros > 0 && lc <= 220.0f) ? lc : e_raw;
}

// ---------------- tiny utility kernels ----------------
__global__ void zero_u32_kernel(unsigned* p, int n) {
    int i = blockIdx.x * blockDim.x + threadIdx.x;
    if (i < n) p[i] = 0u;
}

// histogram of destination (col) including self loops
__global__ void hist_kernel(const long long* ei, unsigned* counts) {
    int e = blockIdx.x * blockDim.x + threadIdx.x;
    if (e >= NE_TOT) return;
    int col = (e < N_EDGES) ? (int)ei[N_EDGES + e] : (e - N_EDGES);
    atomicAdd(&counts[col], 1u);
}

// single-block exclusive scan over 20000 counters; also deg^-1/2
__global__ void scan_kernel(const unsigned* counts, unsigned* row_ptr,
                            unsigned* cursor, float* dinv) {
    __shared__ unsigned s[512];
    int t = threadIdx.x;
    int beg = t * 40;
    int end = beg + 40; if (end > N_NODES) end = N_NODES;
    unsigned part = 0;
    for (int i = beg; i < end; i++) part += counts[i];
    s[t] = part;
    __syncthreads();
    for (int off = 1; off < 512; off <<= 1) {
        unsigned v = (t >= off) ? s[t - off] : 0u;
        __syncthreads();
        s[t] += v;
        __syncthreads();
    }
    unsigned run = s[t] - part;  // exclusive prefix for this chunk
    for (int i = beg; i < end; i++) {
        unsigned c = counts[i];
        row_ptr[i] = run;
        cursor[i]  = run;
        dinv[i]    = rsqrtf((float)c);   // deg >= 1 (self loop always present)
        run += c;
    }
    if (t == 511) row_ptr[N_NODES] = s[511];
}

__global__ void fill_kernel(const long long* ei, unsigned* cursor, const float* dinv,
                            int* csr_src, float* csr_norm) {
    int e = blockIdx.x * blockDim.x + threadIdx.x;
    if (e >= NE_TOT) return;
    int src, col;
    if (e < N_EDGES) { src = (int)ei[e]; col = (int)ei[N_EDGES + e]; }
    else             { src = col = e - N_EDGES; }
    unsigned slot = atomicAdd(&cursor[col], 1u);
    csr_src[slot]  = src;
    csr_norm[slot] = dinv[src] * dinv[col];
}

// ---------------- sketch propagation (gather, atomic-free) ----------------
// minhash: per-node min over in-edges; values < 2^32 so unsigned min == int64 min
template <bool FIRST>
__global__ void mh_prop_kernel(const void* in_, const unsigned* row_ptr,
                               const int* csr_src, unsigned* out) {
    __shared__ int s_src[128];
    int c = blockIdx.x, t = threadIdx.x;
    unsigned beg = row_ptr[c], end = row_ptr[c + 1];
    unsigned m = 0xFFFFFFFFu;
    for (unsigned base = beg; base < end; base += 128) {
        unsigned rem = end - base;
        int n = rem < 128u ? (int)rem : 128;
        if (t < n) s_src[t] = csr_src[base + t];
        __syncthreads();
        for (int i = 0; i < n; i++) {
            int s = s_src[i];
            unsigned v;
            if (FIRST) v = (unsigned)((const unsigned long long*)in_)[(size_t)s * NPERM + t];
            else       v = ((const unsigned*)in_)[(size_t)s * NPERM + t];
            m = v < m ? v : m;
        }
        __syncthreads();
    }
    out[(size_t)c * NPERM + t] = m;
}

// HLL: per-node register-wise max; ranks <= 57 so bytes; 64 threads x 4B words
template <bool FIRST>
__global__ void hll_prop_kernel(const void* in_, const unsigned* row_ptr,
                                const int* csr_src, unsigned* out) {
    __shared__ int s_src[64];
    int c = blockIdx.x, t = threadIdx.x;
    unsigned beg = row_ptr[c], end = row_ptr[c + 1];
    unsigned m = 0u;
    for (unsigned base = beg; base < end; base += 64) {
        unsigned rem = end - base;
        int n = rem < 64u ? (int)rem : 64;
        if (t < n) s_src[t] = csr_src[base + t];
        __syncthreads();
        for (int i = 0; i < n; i++) {
            int s = s_src[i];
            unsigned v;
            if (FIRST) {
                const int* p = (const int*)in_ + (size_t)s * HLL_M + t * 4;
                v = ((unsigned)p[0] & 255u) | (((unsigned)p[1] & 255u) << 8) |
                    (((unsigned)p[2] & 255u) << 16) | (((unsigned)p[3] & 255u) << 24);
            } else {
                v = ((const unsigned*)in_)[(size_t)s * (HLL_M / 4) + t];
            }
            m = bmax4(m, v);
        }
        __syncthreads();
    }
    out[(size_t)c * (HLL_M / 4) + t] = m;
}

// per-node HLL cardinality, both layers (wave 0 -> layer1, wave 1 -> layer2)
__global__ void cards_kernel(const unsigned* hll1, const unsigned* hll2, float* cards) {
    int c = blockIdx.x;
    int w = threadIdx.x >> 5, lane = threadIdx.x & 31;
    const unsigned* row = ((w == 0) ? hll1 : hll2) + (size_t)c * (HLL_M / 4);
    unsigned v0 = row[lane * 2], v1 = row[lane * 2 + 1];
    float sum = 0.f; int zc = 0;
#pragma unroll
    for (int j = 0; j < 4; j++) {
        int r0 = (v0 >> (8 * j)) & 255; sum += ldexpf(1.0f, -r0); zc += (r0 == 0);
        int r1 = (v1 >> (8 * j)) & 255; sum += ldexpf(1.0f, -r1); zc += (r1 == 0);
    }
#pragma unroll
    for (int off = 16; off > 0; off >>= 1) {
        sum += __shfl_xor(sum, off);
        zc  += __shfl_xor(zc, off);
    }
    if (lane == 0) cards[c * 2 + w] = hll_count_f(sum, zc);
}

// ---------------- WMMA GEMM path ----------------
// Pack W[128x128] f32 into f16 fragment order: idx = e + 16*(lane + 32*(nt + 8*kc))
// B-frag: lane l holds column n = nt*16 + (l&15); element e maps to
// K = (e&7) + (e&8 ? 16 : 0) + 8*(l>>4) + 32*kc   (mirror of the 16-bit A layout)
__global__ void packw_kernel(const float* W, _Float16* out) {
    int i = blockIdx.x * blockDim.x + threadIdx.x;
    if (i >= 4 * 8 * 32 * 16) return;
    int e = i & 15, lane = (i >> 4) & 31, nt = (i >> 9) & 7, kc = i >> 12;
    int half = lane >> 4;
    int n = nt * 16 + (lane & 15);
    int K = (e & 7) + ((e & 8) ? 16 : 0) + 8 * half + 32 * kc;
    out[i] = (_Float16)W[K * CH + n];
}

// load the A fragment for K-chunk kc of row m (16-bit A 16x32 layout)
__device__ __forceinline__ v16h load_a_frag(const float* Hin, int m, int half, int kc) {
    const float* pa = Hin + (size_t)m * CH + 32 * kc + 8 * half;
    float4 fa = *(const float4*)(pa);
    float4 fb = *(const float4*)(pa + 4);
    float4 fc = *(const float4*)(pa + 16);
    float4 fd = *(const float4*)(pa + 20);
    v16h a;
    a[0] = (_Float16)fa.x;  a[1] = (_Float16)fa.y;  a[2] = (_Float16)fa.z;  a[3] = (_Float16)fa.w;
    a[4] = (_Float16)fb.x;  a[5] = (_Float16)fb.y;  a[6] = (_Float16)fb.z;  a[7] = (_Float16)fb.w;
    a[8] = (_Float16)fc.x;  a[9] = (_Float16)fc.y;  a[10] = (_Float16)fc.z; a[11] = (_Float16)fc.w;
    a[12] = (_Float16)fd.x; a[13] = (_Float16)fd.y; a[14] = (_Float16)fd.z; a[15] = (_Float16)fd.w;
    return a;
}

// HW = Hin[20000x128] @ W (packed); one wave per 16-row strip, all 8 N-tiles
__global__ void gemm_hw_kernel(const float* Hin, const _Float16* Wpack, float* Out) {
    int wave = (blockIdx.x * blockDim.x + threadIdx.x) >> 5;
    if (wave >= N_NODES / 16) return;
    int lane = threadIdx.x & 31;
    int m0 = wave * 16, half = lane >> 4, m = m0 + (lane & 15);
    v8f zero8 = {0.f, 0.f, 0.f, 0.f, 0.f, 0.f, 0.f, 0.f};
    v8f acc[8];
#pragma unroll
    for (int i = 0; i < 8; i++) acc[i] = zero8;
#pragma unroll
    for (int kc = 0; kc < 4; kc++) {
        v16h a = load_a_frag(Hin, m, half, kc);
#pragma unroll
        for (int nt = 0; nt < 8; nt++) {
            v16h b = *(const v16h*)(Wpack + (size_t)(((kc * 8 + nt) * 32 + lane)) * 16);
            acc[nt] = __builtin_amdgcn_wmma_f32_16x16x32_f16(
                false, a, false, b, (short)0, acc[nt], false, false);
        }
    }
#pragma unroll
    for (int v = 0; v < 8; v++) {
        int row = m0 + v + 8 * half;
#pragma unroll
        for (int nt = 0; nt < 8; nt++) {
            int col = nt * 16 + (lane & 15);
            Out[(size_t)row * CH + col] = acc[nt][v];
        }
    }
}

// final: sigmoid( h2 @ lin_w[0:128] (WMMA) + sf_nodes @ lin_w[128:136] + lin_b )
__global__ void gemm_final_kernel(const float* Hin, const _Float16* Wpack,
                                  const float* sf_nodes, const float* lin_w,
                                  const float* lin_b, float* Out) {
    int wave = (blockIdx.x * blockDim.x + threadIdx.x) >> 5;
    if (wave >= N_NODES / 16) return;
    int lane = threadIdx.x & 31;
    int m0 = wave * 16, half = lane >> 4, m = m0 + (lane & 15);
    v8f zero8 = {0.f, 0.f, 0.f, 0.f, 0.f, 0.f, 0.f, 0.f};
    v8f acc[8];
#pragma unroll
    for (int i = 0; i < 8; i++) acc[i] = zero8;
#pragma unroll
    for (int kc = 0; kc < 4; kc++) {
        v16h a = load_a_frag(Hin, m, half, kc);
#pragma unroll
        for (int nt = 0; nt < 8; nt++) {
            v16h b = *(const v16h*)(Wpack + (size_t)(((kc * 8 + nt) * 32 + lane)) * 16);
            acc[nt] = __builtin_amdgcn_wmma_f32_16x16x32_f16(
                false, a, false, b, (short)0, acc[nt], false, false);
        }
    }
    const float* tail = lin_w + 128 * CH;   // rows 128..135 (sf_nodes part)
#pragma unroll
    for (int v = 0; v < 8; v++) {
        int row = m0 + v + 8 * half;
        float sn[8];
#pragma unroll
        for (int t = 0; t < 8; t++) sn[t] = sf_nodes[(size_t)row * 8 + t];
#pragma unroll
        for (int nt = 0; nt < 8; nt++) {
            int col = nt * 16 + (lane & 15);
            float s = acc[nt][v];
#pragma unroll
            for (int t = 0; t < 8; t++) s += sn[t] * tail[t * CH + col];
            s += lin_b[col];
            Out[(size_t)row * CH + col] = 1.0f / (1.0f + expf(-s));
        }
    }
}

// GCN aggregate: h_out[c][j] = relu( b[j] + sum_in norm * HW[src][j] )
__global__ void aggregate_kernel(const float* HW, const unsigned* row_ptr,
                                 const int* csr_src, const float* csr_norm,
                                 const float* bias, float* Hout) {
    __shared__ int   s_src[128];
    __shared__ float s_nrm[128];
    int c = blockIdx.x, j = threadIdx.x;
    unsigned beg = row_ptr[c], end = row_ptr[c + 1];
    float acc = 0.f;
    for (unsigned base = beg; base < end; base += 128) {
        unsigned rem = end - base;
        int n = rem < 128u ? (int)rem : 128;
        if (j < n) { s_src[j] = csr_src[base + j]; s_nrm[j] = csr_norm[base + j]; }
        __syncthreads();
        for (int i = 0; i < n; i++) acc += s_nrm[i] * HW[(size_t)s_src[i] * CH + j];
        __syncthreads();
    }
    Hout[(size_t)c * CH + j] = fmaxf(acc + bias[j], 0.f);
}

// ---------------- edge pairwise features (one wave per edge) ----------------
__global__ void edge_kernel(const long long* ei, const unsigned* mh1, const unsigned* mh2,
                            const unsigned* hll1, const unsigned* hll2, const float* cards,
                            const float* label_w, const float* label_b,
                            float* sf_pre, double* bn_sum, double* bn_sumsq) {
    __shared__ float s_sum[8], s_sq[8];
    int tid = threadIdx.x;
    if (tid < 8) { s_sum[tid] = 0.f; s_sq[tid] = 0.f; }
    __syncthreads();
    int w = tid >> 5, lane = tid & 31;
    int e = blockIdx.x * 8 + w;               // grid is exact: N_EDGES/8 blocks
    int src = (int)ei[e], dst = (int)ei[N_EDGES + e];

    // Jaccard match counts over 128 permutations (4 perms per lane)
    int c11 = 0, c21 = 0, c12 = 0, c22 = 0;
    const unsigned* m1s = mh1 + (size_t)src * NPERM;
    const unsigned* m2s = mh2 + (size_t)src * NPERM;
    const unsigned* m1d = mh1 + (size_t)dst * NPERM;
    const unsigned* m2d = mh2 + (size_t)dst * NPERM;
#pragma unroll
    for (int q = 0; q < 4; q++) {
        int p = lane + 32 * q;
        unsigned a1 = m1s[p], a2 = m2s[p], b1 = m1d[p], b2 = m2d[p];
        c11 += (a1 == b1); c21 += (a2 == b1); c12 += (a1 == b2); c22 += (a2 == b2);
    }

    // HLL unions over 256 byte-registers (8 bytes per lane)
    const unsigned* r1s = hll1 + (size_t)src * 64;
    const unsigned* r2s = hll2 + (size_t)src * 64;
    const unsigned* r1d = hll1 + (size_t)dst * 64;
    const unsigned* r2d = hll2 + (size_t)dst * 64;
    unsigned ws1[2] = {r1s[lane * 2], r1s[lane * 2 + 1]};
    unsigned ws2[2] = {r2s[lane * 2], r2s[lane * 2 + 1]};
    unsigned wd1[2] = {r1d[lane * 2], r1d[lane * 2 + 1]};
    unsigned wd2[2] = {r2d[lane * 2], r2d[lane * 2 + 1]};
    float su[4] = {0.f, 0.f, 0.f, 0.f};
    int   zc[4] = {0, 0, 0, 0};
#pragma unroll
    for (int wx = 0; wx < 2; wx++) {
#pragma unroll
        for (int j = 0; j < 4; j++) {
            int x1 = (ws1[wx] >> (8 * j)) & 255, x2 = (ws2[wx] >> (8 * j)) & 255;
            int y1 = (wd1[wx] >> (8 * j)) & 255, y2 = (wd2[wx] >> (8 * j)) & 255;
            int u;
            u = x1 > y1 ? x1 : y1; su[0] += ldexpf(1.f, -u); zc[0] += (u == 0);
            u = x2 > y1 ? x2 : y1; su[1] += ldexpf(1.f, -u); zc[1] += (u == 0);
            u = x1 > y2 ? x1 : y2; su[2] += ldexpf(1.f, -u); zc[2] += (u == 0);
            u = x2 > y2 ? x2 : y2; su[3] += ldexpf(1.f, -u); zc[3] += (u == 0);
        }
    }
#pragma unroll
    for (int off = 16; off > 0; off >>= 1) {
        c11 += __shfl_xor(c11, off); c21 += __shfl_xor(c21, off);
        c12 += __shfl_xor(c12, off); c22 += __shfl_xor(c22, off);
#pragma unroll
        for (int k = 0; k < 4; k++) {
            su[k] += __shfl_xor(su[k], off);
            zc[k] += __shfl_xor(zc[k], off);
        }
    }
    if (lane == 0) {
        float pf11 = (c11 * (1.f / 128.f)) * hll_count_f(su[0], zc[0]);
        float pf21 = (c21 * (1.f / 128.f)) * hll_count_f(su[1], zc[1]);
        float pf12 = (c12 * (1.f / 128.f)) * hll_count_f(su[2], zc[2]);
        float pf22 = (c22 * (1.f / 128.f)) * hll_count_f(su[3], zc[3]);
        float f0 = pf11, f1 = pf21 - f0, f2 = pf12 - f0, f3 = pf22 - f0 - f1 - f2;
        float cd0 = cards[dst * 2 + 0], cd1 = cards[dst * 2 + 1];
        float cs0 = cards[src * 2 + 0], cs1 = cards[src * 2 + 1];
        float f4 = cd0 - f0 - f2, f5 = cs0 - f0 - f1;
        float s4 = f0 + f1 + f2 + f3;
        float f6 = cd1 - s4 - f4, f7 = cs1 - s4 - f5;
        float sf[8] = {f0, f1, f2, f3, f4, f5, f6, f7};
#pragma unroll
        for (int i = 0; i < 8; i++) sf[i] = fmaxf(sf[i], 0.f);
#pragma unroll
        for (int j = 0; j < 8; j++) {
            float y = label_b[j];
#pragma unroll
            for (int i = 0; i < 8; i++) y += sf[i] * label_w[i * 8 + j];
            sf_pre[(size_t)e * 8 + j] = y;
            atomicAdd(&s_sum[j], y);
            atomicAdd(&s_sq[j], y * y);
        }
    }
    __syncthreads();
    if (tid < 8)       atomicAdd(&bn_sum[tid], (double)s_sum[tid]);
    else if (tid < 16) atomicAdd(&bn_sumsq[tid - 8], (double)s_sq[tid - 8]);
}

__global__ void bn_finalize_kernel(const double* bn_sum, const double* bn_sumsq,
                                   const float* g, const float* b, float* bnA, float* bnB) {
    int j = threadIdx.x;
    if (j >= 8) return;
    double mu  = bn_sum[j] / (double)N_EDGES;
    double var = bn_sumsq[j] / (double)N_EDGES - mu * mu;
    float inv = (float)(1.0 / sqrt(var + 1e-5));
    float A = g[j] * inv;
    bnA[j] = A;
    bnB[j] = b[j] - (float)mu * A;
}

__global__ void sf_scatter_kernel(const long long* ei, const float* sf_pre,
                                  const float* bnA, const float* bnB, float* sf_nodes) {
    int i = blockIdx.x * blockDim.x + threadIdx.x;
    if (i >= N_EDGES * 8) return;
    int e = i >> 3, j = i & 7;
    int dst = (int)ei[N_EDGES + e];
    float v = fmaxf(sf_pre[i] * bnA[j] + bnB[j], 0.f);
    atomicAdd(&sf_nodes[(size_t)dst * 8 + j], v);
}

// ---------------- host orchestration ----------------
extern "C" void kernel_launch(void* const* d_in, const int* in_sizes, int n_in,
                              void* d_out, int out_size, void* d_ws, size_t ws_size,
                              hipStream_t stream) {
    (void)in_sizes; (void)n_in; (void)out_size; (void)ws_size;
    const float* embed_w = (const float*)d_in[1];
    const float* conv0_w = (const float*)d_in[2];
    const float* conv0_b = (const float*)d_in[3];
    const float* conv1_w = (const float*)d_in[4];
    const float* conv1_b = (const float*)d_in[5];
    const float* label_w = (const float*)d_in[6];
    const float* label_b = (const float*)d_in[7];
    const float* bn_g    = (const float*)d_in[8];
    const float* bn_b    = (const float*)d_in[9];
    const float* lin_w   = (const float*)d_in[10];
    const float* lin_b   = (const float*)d_in[11];
    const long long* ei  = (const long long*)d_in[12];
    const void* init_mh  = d_in[13];   // int64 minhash (values < 2^32)
    const void* init_hll = d_in[14];   // int32 HLL registers
    float* out = (float*)d_out;

    char* base = (char*)d_ws;
    size_t off = 0;
    auto alloc = [&](size_t bytes) -> void* {
        off = (off + 255) & ~(size_t)255;
        void* p = base + off;
        off += bytes;
        return p;
    };
    unsigned* counts   = (unsigned*)alloc((size_t)N_NODES * 4);
    unsigned* row_ptr  = (unsigned*)alloc((size_t)(N_NODES + 1) * 4);
    unsigned* cursor   = (unsigned*)alloc((size_t)N_NODES * 4);
    float*    dinv     = (float*)alloc((size_t)N_NODES * 4);
    int*      csr_src  = (int*)alloc((size_t)NE_TOT * 4);
    float*    csr_norm = (float*)alloc((size_t)NE_TOT * 4);
    unsigned* mh1      = (unsigned*)alloc((size_t)N_NODES * NPERM * 4);
    unsigned* mh2      = (unsigned*)alloc((size_t)N_NODES * NPERM * 4);
    unsigned* hll1     = (unsigned*)alloc((size_t)N_NODES * HLL_M);   // u8 bytes
    unsigned* hll2     = (unsigned*)alloc((size_t)N_NODES * HLL_M);
    float*    cards    = (float*)alloc((size_t)N_NODES * 2 * 4);
    float*    HW       = (float*)alloc((size_t)N_NODES * CH * 4);
    float*    h1       = (float*)alloc((size_t)N_NODES * CH * 4);
    float*    h2       = (float*)alloc((size_t)N_NODES * CH * 4);
    _Float16* wp0      = (_Float16*)alloc(16384 * 2);
    _Float16* wp1      = (_Float16*)alloc(16384 * 2);
    _Float16* wpl      = (_Float16*)alloc(16384 * 2);
    float*    sf_pre   = (float*)alloc((size_t)N_EDGES * 8 * 4);
    float*    sf_nodes = (float*)alloc((size_t)N_NODES * 8 * 4);
    double*   bn_sum   = (double*)alloc(8 * 8);
    double*   bn_sumsq = (double*)alloc(8 * 8);
    float*    bnA      = (float*)alloc(32);
    float*    bnB      = (float*)alloc(32);

    // init
    zero_u32_kernel<<<(N_NODES + 255) / 256, 256, 0, stream>>>(counts, N_NODES);
    zero_u32_kernel<<<(N_NODES * 8 + 255) / 256, 256, 0, stream>>>((unsigned*)sf_nodes, N_NODES * 8);
    zero_u32_kernel<<<1, 32, 0, stream>>>((unsigned*)bn_sum, 16);
    zero_u32_kernel<<<1, 32, 0, stream>>>((unsigned*)bn_sumsq, 16);

    // CSR build (dest-major, self loops appended)
    hist_kernel<<<(NE_TOT + 255) / 256, 256, 0, stream>>>(ei, counts);
    scan_kernel<<<1, 512, 0, stream>>>(counts, row_ptr, cursor, dinv);
    fill_kernel<<<(NE_TOT + 255) / 256, 256, 0, stream>>>(ei, cursor, dinv, csr_src, csr_norm);

    // sketch propagation (2 hops)
    mh_prop_kernel<true><<<N_NODES, 128, 0, stream>>>(init_mh, row_ptr, csr_src, mh1);
    mh_prop_kernel<false><<<N_NODES, 128, 0, stream>>>(mh1, row_ptr, csr_src, mh2);
    hll_prop_kernel<true><<<N_NODES, 64, 0, stream>>>(init_hll, row_ptr, csr_src, hll1);
    hll_prop_kernel<false><<<N_NODES, 64, 0, stream>>>(hll1, row_ptr, csr_src, hll2);
    cards_kernel<<<N_NODES, 64, 0, stream>>>(hll1, hll2, cards);

    // GCN layer 0:  HW = embed_w @ W0 (WMMA), aggregate + bias + relu
    packw_kernel<<<64, 256, 0, stream>>>(conv0_w, wp0);
    gemm_hw_kernel<<<313, 128, 0, stream>>>(embed_w, wp0, HW);
    aggregate_kernel<<<N_NODES, 128, 0, stream>>>(HW, row_ptr, csr_src, csr_norm, conv0_b, h1);
    // GCN layer 1
    packw_kernel<<<64, 256, 0, stream>>>(conv1_w, wp1);
    gemm_hw_kernel<<<313, 128, 0, stream>>>(h1, wp1, HW);
    aggregate_kernel<<<N_NODES, 128, 0, stream>>>(HW, row_ptr, csr_src, csr_norm, conv1_b, h2);

    // pairwise subgraph features + label linear + BN stats
    edge_kernel<<<N_EDGES / 8, 256, 0, stream>>>(ei, mh1, mh2, hll1, hll2, cards,
                                                 label_w, label_b, sf_pre, bn_sum, bn_sumsq);
    bn_finalize_kernel<<<1, 8, 0, stream>>>(bn_sum, bn_sumsq, bn_g, bn_b, bnA, bnB);
    sf_scatter_kernel<<<(N_EDGES * 8 + 255) / 256, 256, 0, stream>>>(ei, sf_pre, bnA, bnB, sf_nodes);

    // final linear (WMMA for the 128-wide part) + sigmoid
    packw_kernel<<<64, 256, 0, stream>>>(lin_w, wpl);
    gemm_final_kernel<<<313, 128, 0, stream>>>(h2, wpl, sf_nodes, lin_w, lin_b, out);
}
